// GQA_44289702756785
// MI455X (gfx1250) — compile-verified, matrix operand order checked by reading, not run on hardware
//
#include <hip/hip_runtime.h>
#include <cstdint>
#include <cstddef>

typedef __attribute__((ext_vector_type(16))) __bf16 v16bf;
typedef __attribute__((ext_vector_type(8)))  __bf16 v8bf;
typedef __attribute__((ext_vector_type(8)))  float  v8f;

#define BDIM  2
#define TDIM  2048
#define NEMB  2048
#define NH    16
#define NKV   4
#define HD    128
#define REP   4     // NH / NKV

union Frag { v16bf v; v8bf h2[2]; };

// ---------------------------------------------------------------------------
// f32 -> bf16 convert (8 elements / thread)
// ---------------------------------------------------------------------------
__global__ void cvt_bf16x8(const float* __restrict__ in, __bf16* __restrict__ out, int n8) {
    int i = blockIdx.x * blockDim.x + threadIdx.x;
    if (i >= n8) return;
    const float* p = in + (size_t)i * 8;
    v8bf o;
#pragma unroll
    for (int e = 0; e < 8; ++e) o[e] = (__bf16)p[e];
    *(v8bf*)(out + (size_t)i * 8) = o;
}

// ---------------------------------------------------------------------------
// Generic bf16 WMMA GEMM:  C[M,N] = A[M,K] @ W[N,K]^T + bias
//   mode 0: write f32 row-major [M][N]            (final projection -> d_out)
//   mode 1: RoPE + write bf16 head-major [B,H,T,HD]   (Q, K)
//   mode 2: write bf16 head-major [B,H,T,HD]          (V)
// Block = 128 threads (4 waves, 2x2), block tile 128x128, wave tile 64x64.
// ---------------------------------------------------------------------------
__global__ __launch_bounds__(128)
void gemm_wmma(const __bf16* __restrict__ A, const __bf16* __restrict__ W,
               const float* __restrict__ bias, float* __restrict__ outF,
               __bf16* __restrict__ outH, int M, int N, int K, int mode)
{
    const int lane  = threadIdx.x & 31;
    const int wave  = threadIdx.x >> 5;
    const int lrow  = lane & 15;
    const int lhalf = lane >> 4;
    const int m0 = blockIdx.y * 128 + (wave >> 1) * 64;
    const int n0 = blockIdx.x * 128 + (wave & 1) * 64;

    v8f acc[4][4];
    const v8f vzero = {0.f,0.f,0.f,0.f,0.f,0.f,0.f,0.f};
#pragma unroll
    for (int i = 0; i < 4; ++i)
#pragma unroll
        for (int j = 0; j < 4; ++j) acc[i][j] = vzero;

    for (int kk = 0; kk < K; kk += 32) {
        Frag a[4], b[4];
#pragma unroll
        for (int i = 0; i < 4; ++i) {
            const __bf16* p = A + (size_t)(m0 + i * 16 + lrow) * K + kk + lhalf * 8;
            a[i].h2[0] = *(const v8bf*)p;
            a[i].h2[1] = *(const v8bf*)(p + 16);
        }
#pragma unroll
        for (int j = 0; j < 4; ++j) {
            const __bf16* p = W + (size_t)(n0 + j * 16 + lrow) * K + kk + lhalf * 8;
            b[j].h2[0] = *(const v8bf*)p;
            b[j].h2[1] = *(const v8bf*)(p + 16);
        }
        if (kk + 32 < K) {  // global_prefetch_b8 of next K-slab
            __builtin_prefetch(A + (size_t)(m0 + lrow) * K + kk + 32, 0, 1);
            __builtin_prefetch(W + (size_t)(n0 + lrow) * K + kk + 32, 0, 1);
        }
#pragma unroll
        for (int i = 0; i < 4; ++i)
#pragma unroll
            for (int j = 0; j < 4; ++j)
                acc[i][j] = __builtin_amdgcn_wmma_f32_16x16x32_bf16(
                    false, a[i].v, false, b[j].v, (short)0, acc[i][j], false, false);
    }

    // Epilogue: bias (+ RoPE) + layout
#pragma unroll
    for (int j = 0; j < 4; ++j) {
        const int col = n0 + j * 16 + lrow;
        const float bv = bias[col];
#pragma unroll
        for (int i = 0; i < 4; ++i) {
            const int rowbase = m0 + i * 16 + lhalf * 8;
#pragma unroll
            for (int r = 0; r < 8; ++r) {
                const int row = rowbase + r;
                float val = acc[i][j][r] + bv;
                if (mode == 0) {
                    outF[(size_t)row * N + col] = val;
                } else {
                    if (mode == 1) {
                        const int d  = col & (HD - 1);
                        const int pi = d >> 1;
                        // inv_freq = 10000^(-2*pi/128); ln(10000)=9.2103403...
                        const float freq = __expf(-(float)(2 * pi) * (9.210340371976184f / 128.0f));
                        const float t = (float)(row & (TDIM - 1));
                        float sn, cs;
                        sincosf(t * freq, &sn, &cs);
                        const float other = __shfl_xor(val, 1);   // pair partner (even<->odd)
                        val = (d & 1) ? (other * sn + val * cs)   // odd' = x1*sin + x2*cos
                                      : (val * cs - other * sn);  // even'= x1*cos - x2*sin
                    }
                    const int h = col >> 7, d = col & 127;
                    const int bb = row / TDIM, tt = row % TDIM;
                    const int Hn = N >> 7;
                    outH[(((size_t)bb * Hn + h) * TDIM + tt) * HD + d] = (__bf16)val;
                }
            }
        }
    }
}

// ---------------------------------------------------------------------------
// Flash-style causal GQA attention.
// Grid: (T/64, NH, B). Block: 128 threads = 4 waves, each owns 16 q-rows.
// K-tile (32x128) and transposed V-tile staged in LDS (shared KV head x4 reuse).
// Online softmax; scale E^-1/2 applied post-softmax (per reference).
// ---------------------------------------------------------------------------
__global__ __launch_bounds__(128)
void attn_wmma(const __bf16* __restrict__ Q, const __bf16* __restrict__ Kh,
               const __bf16* __restrict__ Vh, __bf16* __restrict__ AO)
{
    __shared__ __bf16 Ks[32 * HD];      // [k][d]
    __shared__ __bf16 Vs[HD * 32];      // transposed: [d][k]
    __shared__ __bf16 Ps[4][16 * 32];   // per-wave P staging [q][k]

    const int tid   = threadIdx.x;
    const int lane  = tid & 31, wave = tid >> 5;
    const int lrow  = lane & 15, lhalf = lane >> 4;
    const int b = blockIdx.z, h = blockIdx.y;
    const int kvh   = h / REP;
    const int qbase = blockIdx.x * 64;
    const int q0    = qbase + wave * 16;

    // Q fragments for this wave's 16 rows (reused for all k-tiles)
    const __bf16* Qb = Q + ((size_t)(b * NH + h) * TDIM) * HD;
    Frag qf[4];
    {
        const __bf16* p = Qb + (size_t)(q0 + lrow) * HD + lhalf * 8;
#pragma unroll
        for (int dt = 0; dt < 4; ++dt) {
            qf[dt].h2[0] = *(const v8bf*)(p + dt * 32);
            qf[dt].h2[1] = *(const v8bf*)(p + dt * 32 + 16);
        }
    }

    v8f acc[8];
    const v8f vzero = {0.f,0.f,0.f,0.f,0.f,0.f,0.f,0.f};
#pragma unroll
    for (int dt = 0; dt < 8; ++dt) acc[dt] = vzero;
    float mr[8], lr[8];
#pragma unroll
    for (int r = 0; r < 8; ++r) { mr[r] = -__builtin_inff(); lr[r] = 0.f; }

    const __bf16* Kb = Kh + ((size_t)(b * NKV + kvh) * TDIM) * HD;
    const __bf16* Vb = Vh + ((size_t)(b * NKV + kvh) * TDIM) * HD;

    const int nkt = qbase / 32 + 2;   // causal: k <= qbase+63
    for (int kt = 0; kt < nkt; ++kt) {
        const int kb = kt * 32;
        {   // cooperative staging: thread = (k-row, 32-wide d chunk)
            const int kkr = tid & 31, c = tid >> 5;
            const __bf16* ks = Kb + (size_t)(kb + kkr) * HD + c * 32;
            v8bf* dK = (v8bf*)&Ks[kkr * HD + c * 32];
#pragma unroll
            for (int e = 0; e < 4; ++e) dK[e] = *(const v8bf*)(ks + e * 8);
            const __bf16* vs = Vb + (size_t)(kb + kkr) * HD + c * 32;
#pragma unroll
            for (int e = 0; e < 32; ++e) Vs[(c * 32 + e) * 32 + kkr] = vs[e];
        }
        __syncthreads();

        if (kb <= q0 + 15) {   // wave-uniform -> EXEC all-ones inside (WMMA-legal)
            // S = Q Kt : two 16x16 score tiles over 32 keys, K-dim = 128
            v8f s0 = vzero, s1 = vzero;
#pragma unroll
            for (int dt = 0; dt < 4; ++dt) {
                Frag kf0, kf1;
                const __bf16* p0 = &Ks[lrow * HD + dt * 32 + lhalf * 8];
                kf0.h2[0] = *(const v8bf*)p0; kf0.h2[1] = *(const v8bf*)(p0 + 16);
                const __bf16* p1 = &Ks[(16 + lrow) * HD + dt * 32 + lhalf * 8];
                kf1.h2[0] = *(const v8bf*)p1; kf1.h2[1] = *(const v8bf*)(p1 + 16);
                s0 = __builtin_amdgcn_wmma_f32_16x16x32_bf16(false, qf[dt].v, false, kf0.v, (short)0, s0, false, false);
                s1 = __builtin_amdgcn_wmma_f32_16x16x32_bf16(false, qf[dt].v, false, kf1.v, (short)0, s1, false, false);
            }
            // causal mask + online softmax (row = 16-lane group reduction)
            __bf16* ps = Ps[wave];
#pragma unroll
            for (int r = 0; r < 8; ++r) {
                const int qr = q0 + r + lhalf * 8;
                float v0 = s0[r]; if (kb + lrow      > qr) v0 = -__builtin_inff();
                float v1 = s1[r]; if (kb + 16 + lrow > qr) v1 = -__builtin_inff();
                float mx = fmaxf(v0, v1);
#pragma unroll
                for (int off = 1; off <= 8; off <<= 1) mx = fmaxf(mx, __shfl_xor(mx, off));
                const float mnew = fmaxf(mr[r], mx);
                const float corr = __expf(mr[r] - mnew);
                mr[r] = mnew;
                const float p0 = __expf(v0 - mnew);
                const float p1 = __expf(v1 - mnew);
                float rs = p0 + p1;
#pragma unroll
                for (int off = 1; off <= 8; off <<= 1) rs += __shfl_xor(rs, off);
                lr[r] = lr[r] * corr + rs;
#pragma unroll
                for (int dt = 0; dt < 8; ++dt) acc[dt][r] *= corr;
                ps[(r + lhalf * 8) * 32 + lrow]      = (__bf16)p0;
                ps[(r + lhalf * 8) * 32 + 16 + lrow] = (__bf16)p1;
            }
            // P (A-frag, same-wave LDS round-trip) x V (B-frag from transposed tile)
            Frag pf;
            const __bf16* pp = &Ps[wave][lrow * 32 + lhalf * 8];
            pf.h2[0] = *(const v8bf*)pp; pf.h2[1] = *(const v8bf*)(pp + 16);
#pragma unroll
            for (int dt = 0; dt < 8; ++dt) {
                Frag vf;
                const __bf16* vp = &Vs[(dt * 16 + lrow) * 32 + lhalf * 8];
                vf.h2[0] = *(const v8bf*)vp; vf.h2[1] = *(const v8bf*)(vp + 16);
                acc[dt] = __builtin_amdgcn_wmma_f32_16x16x32_bf16(false, pf.v, false, vf.v, (short)0, acc[dt], false, false);
            }
        }
        __syncthreads();
    }

    // normalize, apply E^-1/2 (post-softmax per reference), write [B,T,E] bf16
    const float osc = 0.02209708691207961f;  // 2048^-0.5
#pragma unroll
    for (int r = 0; r < 8; ++r) {
        const int qr = q0 + r + lhalf * 8;
        const float inv = osc / lr[r];
        const size_t base = ((size_t)b * TDIM + qr) * NEMB + h * HD;
#pragma unroll
        for (int dt = 0; dt < 8; ++dt)
            AO[base + dt * 16 + lrow] = (__bf16)(acc[dt][r] * inv);
    }
}

// ---------------------------------------------------------------------------
extern "C" void kernel_launch(void* const* d_in, const int* in_sizes, int n_in,
                              void* d_out, int out_size, void* d_ws, size_t ws_size,
                              hipStream_t stream) {
    (void)in_sizes; (void)n_in; (void)out_size; (void)ws_size;
    const float* x  = (const float*)d_in[0];
    const float* Wq = (const float*)d_in[1];
    const float* bq = (const float*)d_in[2];
    const float* Wk = (const float*)d_in[3];
    const float* bk = (const float*)d_in[4];
    const float* Wv = (const float*)d_in[5];
    const float* bv = (const float*)d_in[6];
    const float* Wo = (const float*)d_in[7];
    const float* bo = (const float*)d_in[8];
    float* out = (float*)d_out;

    const int M = BDIM * TDIM;        // 4096
    char* ws = (char*)d_ws;
    size_t off = 0;
    auto alloc = [&](size_t bytes) -> void* {
        void* p = ws + off; off += (bytes + 255) & ~(size_t)255; return p;
    };
    __bf16* xb  = (__bf16*)alloc((size_t)M * NEMB * 2);
    __bf16* Wqb = (__bf16*)alloc((size_t)NEMB * NEMB * 2);
    __bf16* Wkb = (__bf16*)alloc((size_t)(NKV * HD) * NEMB * 2);
    __bf16* Wvb = (__bf16*)alloc((size_t)(NKV * HD) * NEMB * 2);
    __bf16* Wob = (__bf16*)alloc((size_t)NEMB * NEMB * 2);
    __bf16* Qr  = (__bf16*)alloc((size_t)BDIM * NH  * TDIM * HD * 2);
    __bf16* Kr  = (__bf16*)alloc((size_t)BDIM * NKV * TDIM * HD * 2);
    __bf16* Vr  = (__bf16*)alloc((size_t)BDIM * NKV * TDIM * HD * 2);
    __bf16* AO  = (__bf16*)alloc((size_t)M * NEMB * 2);

    auto cvt = [&](const float* src, __bf16* dst, size_t n) {
        int n8 = (int)(n / 8);
        cvt_bf16x8<<<(n8 + 255) / 256, 256, 0, stream>>>(src, dst, n8);
    };
    cvt(x,  xb,  (size_t)M * NEMB);
    cvt(Wq, Wqb, (size_t)NEMB * NEMB);
    cvt(Wk, Wkb, (size_t)(NKV * HD) * NEMB);
    cvt(Wv, Wvb, (size_t)(NKV * HD) * NEMB);
    cvt(Wo, Wob, (size_t)NEMB * NEMB);

    // QKV projections (RoPE fused for Q/K; head-major bf16 outputs)
    gemm_wmma<<<dim3(NEMB / 128, M / 128), 128, 0, stream>>>(
        xb, Wqb, bq, nullptr, Qr, M, NEMB, NEMB, 1);
    gemm_wmma<<<dim3((NKV * HD) / 128, M / 128), 128, 0, stream>>>(
        xb, Wkb, bk, nullptr, Kr, M, NKV * HD, NEMB, 1);
    gemm_wmma<<<dim3((NKV * HD) / 128, M / 128), 128, 0, stream>>>(
        xb, Wvb, bv, nullptr, Vr, M, NKV * HD, NEMB, 2);

    // Causal GQA attention
    attn_wmma<<<dim3(TDIM / 64, NH, BDIM), 128, 0, stream>>>(Qr, Kr, Vr, AO);

    // Output projection -> f32 d_out
    gemm_wmma<<<dim3(NEMB / 128, M / 128), 128, 0, stream>>>(
        AO, Wob, bo, out, nullptr, M, NEMB, NEMB, 0);
}
